// tPCN_75342316306993
// MI455X (gfx1250) — compile-verified, matrix operand order checked by reading, not run on hardware
//
#include <hip/hip_runtime.h>
#include <math.h>

#define B_DIM 4096
#define H_DIM 1024
#define NOBS  128
#define NDIR  64
#define MAX_ITER 10
#define LR_ITER 1e-3f
#define TOL 1e-3f

typedef __bf16 bf16x8  __attribute__((ext_vector_type(8)));
typedef __bf16 bf16x16 __attribute__((ext_vector_type(16)));
typedef float  f32x8   __attribute__((ext_vector_type(8)));

#define LS 40  // LDS row pitch (bf16): 80B rows -> conflict-free ds_load_b128 frags

#define USE_ASYNC 1   // GLOBAL_LOAD_ASYNC_TO_LDS_B128 via inline asm (ASYNCcnt-tracked)

// 16-byte global->LDS copy; async DMA path on gfx1250.
__device__ __forceinline__ void cp16(__bf16* l, const __bf16* g) {
#if USE_ASYNC
  unsigned lds_off = (unsigned)(size_t)(const __attribute__((address_space(3))) void*)l;
  asm volatile("global_load_async_to_lds_b128 %0, %1, off"
               :: "v"(lds_off), "v"(g) : "memory");
#else
  *(bf16x8*)l = *(const bf16x8*)g;
#endif
}

__device__ __forceinline__ void wait_stage() {
#if USE_ASYNC
  asm volatile("s_wait_asynccnt 0x0" ::: "memory");
#endif
  __syncthreads();
}

// Fragment load matching ISA 16-bit A/B layout:
// lane<16: elems0..7=K 0..7, elems8..15=K 16..23 ; lane>=16: +8 on K base.
__device__ __forceinline__ bf16x16 frag_ld(const __bf16* base, int lane) {
  const int r  = lane & 15;
  const int kb = (lane >> 4) << 3;
  bf16x8 a = *(const bf16x8*)(base + r * LS + kb);
  bf16x8 b = *(const bf16x8*)(base + r * LS + kb + 16);
  return __builtin_shufflevector(a, b, 0,1,2,3,4,5,6,7,8,9,10,11,12,13,14,15);
}

#define WMMA_BF16(A, Bm, Cc) \
  __builtin_amdgcn_wmma_f32_16x16x32_bf16(false, A, false, Bm, (short)0, Cc, false, false)

// C[M,N] = (Ahi+Alo) @ (Bhi+Blo)^T, bf16x3 error-corrected, f32 accumulate.
// A,[M,K] and B,[N,K] row-major bf16 hi/lo. Block tile 128x64, wave tile 32x32.
__global__ __launch_bounds__(256) void gemm_bf16x3(
    const __bf16* __restrict__ Ahi, const __bf16* __restrict__ Alo, int lda,
    const __bf16* __restrict__ Bhi, const __bf16* __restrict__ Blo, int ldb,
    float* __restrict__ C, int ldc,
    const float* __restrict__ drive, int do_relu, int K)
{
  __shared__ __align__(16) __bf16 AsHi[128 * LS];
  __shared__ __align__(16) __bf16 AsLo[128 * LS];
  __shared__ __align__(16) __bf16 BsHi[64 * LS];
  __shared__ __align__(16) __bf16 BsLo[64 * LS];

  const int tid  = threadIdx.x;
  const int lane = tid & 31;
  const int wave = tid >> 5;
  const int wm = wave >> 1;            // 0..3 -> 32-row M sub-block
  const int wn = wave & 1;             // 0..1 -> 32-col N sub-block
  const int m0 = blockIdx.y * 128;
  const int n0 = blockIdx.x * 64;

  // staging maps (all 16B chunks, 16B aligned in padded LDS rows)
  const int ra = tid >> 1, ca = (tid & 1) << 4;   // A: 128 rows x 32 cols, 32 elems/thread
  const int rb = tid >> 2, cb = (tid & 3) << 3;   // B:  64 rows x 32 cols, 8+8 elems/thread

  f32x8 acc00 = {}, acc01 = {}, acc10 = {}, acc11 = {};

  for (int k0 = 0; k0 < K; k0 += 32) {
    const __bf16* gAh = Ahi + (size_t)(m0 + ra) * lda + k0 + ca;
    const __bf16* gAl = Alo + (size_t)(m0 + ra) * lda + k0 + ca;
    __bf16* sAh = &AsHi[ra * LS + ca];
    __bf16* sAl = &AsLo[ra * LS + ca];
    cp16(sAh, gAh); cp16(sAh + 8, gAh + 8);
    cp16(sAl, gAl); cp16(sAl + 8, gAl + 8);
    cp16(&BsHi[rb * LS + cb], Bhi + (size_t)(n0 + rb) * ldb + k0 + cb);
    cp16(&BsLo[rb * LS + cb], Blo + (size_t)(n0 + rb) * ldb + k0 + cb);
#if !USE_ASYNC
    if (k0 + 32 < K) __builtin_prefetch(gAh + 32, 0, 1);
#endif
    wait_stage();

    bf16x16 a0h = frag_ld(AsHi + (wm * 32) * LS, lane);
    bf16x16 a0l = frag_ld(AsLo + (wm * 32) * LS, lane);
    bf16x16 a1h = frag_ld(AsHi + (wm * 32 + 16) * LS, lane);
    bf16x16 a1l = frag_ld(AsLo + (wm * 32 + 16) * LS, lane);
    bf16x16 b0h = frag_ld(BsHi + (wn * 32) * LS, lane);
    bf16x16 b0l = frag_ld(BsLo + (wn * 32) * LS, lane);
    bf16x16 b1h = frag_ld(BsHi + (wn * 32 + 16) * LS, lane);
    bf16x16 b1l = frag_ld(BsLo + (wn * 32 + 16) * LS, lane);

    acc00 = WMMA_BF16(a0h, b0h, acc00);
    acc00 = WMMA_BF16(a0h, b0l, acc00);
    acc00 = WMMA_BF16(a0l, b0h, acc00);
    acc01 = WMMA_BF16(a0h, b1h, acc01);
    acc01 = WMMA_BF16(a0h, b1l, acc01);
    acc01 = WMMA_BF16(a0l, b1h, acc01);
    acc10 = WMMA_BF16(a1h, b0h, acc10);
    acc10 = WMMA_BF16(a1h, b0l, acc10);
    acc10 = WMMA_BF16(a1l, b0h, acc10);
    acc11 = WMMA_BF16(a1h, b1h, acc11);
    acc11 = WMMA_BF16(a1h, b1l, acc11);
    acc11 = WMMA_BF16(a1l, b1h, acc11);
    __syncthreads();
  }

  // C/D layout: n = lane&15, m = r + 8*(lane>>4)
  const int mrow = m0 + wm * 32 + ((lane >> 4) << 3);
  const int ncol = n0 + wn * 32 + (lane & 15);
#pragma unroll
  for (int r = 0; r < 8; ++r) {
    size_t i00 = (size_t)(mrow + r) * ldc + ncol;
    size_t i01 = i00 + 16;
    size_t i10 = (size_t)(mrow + 16 + r) * ldc + ncol;
    size_t i11 = i10 + 16;
    float x00 = acc00[r], x01 = acc01[r], x10 = acc10[r], x11 = acc11[r];
    if (drive) { x00 += drive[i00]; x01 += drive[i01]; x10 += drive[i10]; x11 += drive[i11]; }
    if (do_relu) {
      x00 = fmaxf(x00, 0.f); x01 = fmaxf(x01, 0.f);
      x10 = fmaxf(x10, 0.f); x11 = fmaxf(x11, 0.f);
    }
    C[i00] = x00; C[i01] = x01; C[i10] = x10; C[i11] = x11;
  }
}

// ---------- small helper kernels ----------
__global__ void split_kernel(const float* __restrict__ src, __bf16* hi, __bf16* lo, int n) {
  for (int i = blockIdx.x * blockDim.x + threadIdx.x; i < n; i += gridDim.x * blockDim.x) {
    float x = src[i];
    __bf16 h = (__bf16)x;
    hi[i] = h; lo[i] = (__bf16)(x - (float)h);
  }
}

// WoutT[h][o] = Wout[o][h], split hi/lo
__global__ void splitT_kernel(const float* __restrict__ Wout, __bf16* hi, __bf16* lo, int n) {
  for (int i = blockIdx.x * blockDim.x + threadIdx.x; i < n; i += gridDim.x * blockDim.x) {
    int h = i >> 7, o = i & 127;
    float x = Wout[(size_t)o * H_DIM + h];
    __bf16 hh = (__bf16)x;
    hi[i] = hh; lo[i] = (__bf16)(x - (float)hh);
  }
}

// drive[b,h] = Win[h, dir[b]]   (one-hot matvec == column gather)
__global__ void drive_kernel(const float* __restrict__ Win, const int* __restrict__ dirs,
                             float* __restrict__ drive, int n) {
  for (int i = blockIdx.x * blockDim.x + threadIdx.x; i < n; i += gridDim.x * blockDim.x) {
    int b = i >> 10, h = i & 1023;
    drive[i] = Win[h * NDIR + dirs[b]];
  }
}

__global__ void copy_kernel(const float* __restrict__ s, float* __restrict__ d, int n) {
  for (int i = blockIdx.x * blockDim.x + threadIdx.x; i < n; i += gridDim.x * blockDim.x) d[i] = s[i];
}

__global__ void init_flags_kernel(int* flags) { flags[0] = 0; flags[1] = 0; }

__global__ void reduce_diff_kernel(const float* __restrict__ p, const float* __restrict__ c,
                                   float* __restrict__ part, int n) {
  __shared__ float sm[256];
  float s = 0.f;
  for (int i = blockIdx.x * blockDim.x + threadIdx.x; i < n; i += gridDim.x * blockDim.x)
    s += p[i] - c[i];
  sm[threadIdx.x] = s; __syncthreads();
  for (int o = 128; o; o >>= 1) { if (threadIdx.x < o) sm[threadIdx.x] += sm[threadIdx.x + o]; __syncthreads(); }
  if (!threadIdx.x) part[blockIdx.x] = sm[0];
}

__global__ void flags_kernel(const float* __restrict__ part, int* flags, float invN) {
  __shared__ float sm[256];
  sm[threadIdx.x] = part[threadIdx.x]; __syncthreads();
  for (int o = 128; o; o >>= 1) { if (threadIdx.x < o) sm[threadIdx.x] += sm[threadIdx.x + o]; __syncthreads(); }
  if (!threadIdx.x) {
    int cond = fabsf(sm[0] * invN) > TOL;
    int done = flags[1];
    flags[0] = (!done) && cond;       // active
    flags[1] = done || (!cond);       // done latches
  }
}

__global__ void advance_kernel(float* prev, float* cur, const float* __restrict__ ncur,
                               const int* __restrict__ flags, int n) {
  if (!flags[0]) return;
  for (int i = blockIdx.x * blockDim.x + threadIdx.x; i < n; i += gridDim.x * blockDim.x) {
    prev[i] = cur[i];
    cur[i]  = ncur[i];
  }
}

// one wave per row of 128 logits; Jv = f*(eps - <f,eps>), eps = onehot - f
__global__ __launch_bounds__(256) void softmax_jv_kernel(const float* __restrict__ logits,
                                                         const int* __restrict__ obs,
                                                         float* __restrict__ Jv, int write_jv,
                                                         float* __restrict__ preds) {
  int row  = (blockIdx.x * blockDim.x + threadIdx.x) >> 5;
  int lane = threadIdx.x & 31;
  if (row >= B_DIM) return;
  const float* lg = logits + (size_t)row * NOBS;
  float x[4], mx = -3.4e38f;
#pragma unroll
  for (int j = 0; j < 4; ++j) { x[j] = lg[lane + 32 * j]; mx = fmaxf(mx, x[j]); }
  for (int o = 16; o; o >>= 1) mx = fmaxf(mx, __shfl_xor(mx, o, 32));
  float s = 0.f;
#pragma unroll
  for (int j = 0; j < 4; ++j) { x[j] = expf(x[j] - mx); s += x[j]; }
  for (int o = 16; o; o >>= 1) s += __shfl_xor(s, o, 32);
  float inv = 1.f / s;
  if (!write_jv) {
#pragma unroll
    for (int j = 0; j < 4; ++j) preds[(size_t)row * NOBS + lane + 32 * j] = x[j] * inv;
    return;
  }
  int ob = obs[row];
  float f[4], e[4], s2 = 0.f;
#pragma unroll
  for (int j = 0; j < 4; ++j) {
    f[j] = x[j] * inv;
    e[j] = ((lane + 32 * j) == ob ? 1.f : 0.f) - f[j];
    s2 += f[j] * e[j];
  }
  for (int o = 16; o; o >>= 1) s2 += __shfl_xor(s2, o, 32);
#pragma unroll
  for (int j = 0; j < 4; ++j) Jv[(size_t)row * NOBS + lane + 32 * j] = f[j] * (e[j] - s2);
}

// out = R / (||R_row|| + 1e-6)
__global__ __launch_bounds__(256) void rownorm_kernel(const float* __restrict__ R, float* __restrict__ out) {
  __shared__ float sm[256];
  size_t base = (size_t)blockIdx.x * H_DIM;
  float v[4], ss = 0.f;
#pragma unroll
  for (int j = 0; j < 4; ++j) { v[j] = R[base + threadIdx.x + 256 * j]; ss += v[j] * v[j]; }
  sm[threadIdx.x] = ss; __syncthreads();
  for (int o = 128; o; o >>= 1) { if (threadIdx.x < o) sm[threadIdx.x] += sm[threadIdx.x + o]; __syncthreads(); }
  float inv = 1.f / (sqrtf(sm[0]) + 1e-6f);
#pragma unroll
  for (int j = 0; j < 4; ++j) out[base + threadIdx.x + 256 * j] = v[j] * inv;
}

// ncur = norm(cur + lr*((R - cur) + U))
__global__ __launch_bounds__(256) void combine_norm_kernel(const float* __restrict__ cur,
                                                           const float* __restrict__ R,
                                                           const float* __restrict__ U,
                                                           float* __restrict__ ncur) {
  __shared__ float sm[256];
  size_t base = (size_t)blockIdx.x * H_DIM;
  float v[4], ss = 0.f;
#pragma unroll
  for (int j = 0; j < 4; ++j) {
    size_t i = base + threadIdx.x + 256 * j;
    float c = cur[i];
    float val = c + LR_ITER * ((R[i] - c) + U[i]);
    v[j] = val; ss += val * val;
  }
  sm[threadIdx.x] = ss; __syncthreads();
  for (int o = 128; o; o >>= 1) { if (threadIdx.x < o) sm[threadIdx.x] += sm[threadIdx.x + o]; __syncthreads(); }
  float inv = 1.f / (sqrtf(sm[0]) + 1e-6f);
#pragma unroll
  for (int j = 0; j < 4; ++j) ncur[base + threadIdx.x + 256 * j] = v[j] * inv;
}

// ------------------------------------------------------------------
extern "C" void kernel_launch(void* const* d_in, const int* in_sizes, int n_in,
                              void* d_out, int out_size, void* d_ws, size_t ws_size,
                              hipStream_t stream) {
  const int*   dirs = (const int*)d_in[0];
  const int*   obs  = (const int*)d_in[1];
  const float* Wr   = (const float*)d_in[2];
  const float* Win  = (const float*)d_in[3];
  const float* Wout = (const float*)d_in[4];
  const float* ps   = (const float*)d_in[5];
  float* out = (float*)d_out;

  const int n   = B_DIM * H_DIM;
  const int nBO = B_DIM * NOBS;

  char* w = (char*)d_ws;
  size_t off = 0;
  auto alloc = [&](size_t bytes) { size_t o = off; off += (bytes + 255) & ~(size_t)255; return o; };
  float*  drive  = (float*)(w + alloc((size_t)n * 4));
  float*  prevb  = (float*)(w + alloc((size_t)n * 4));
  float*  curb   = (float*)(w + alloc((size_t)n * 4));
  float*  newc   = (float*)(w + alloc((size_t)n * 4));
  float*  Rbuf   = (float*)(w + alloc((size_t)n * 4));
  float*  Ubuf   = (float*)(w + alloc((size_t)n * 4));
  float*  logits = (float*)(w + alloc((size_t)nBO * 4));
  float*  Jv     = (float*)(w + alloc((size_t)nBO * 4));
  __bf16* WrHi   = (__bf16*)(w + alloc((size_t)H_DIM * H_DIM * 2));
  __bf16* WrLo   = (__bf16*)(w + alloc((size_t)H_DIM * H_DIM * 2));
  __bf16* WoHi   = (__bf16*)(w + alloc((size_t)NOBS * H_DIM * 2));
  __bf16* WoLo   = (__bf16*)(w + alloc((size_t)NOBS * H_DIM * 2));
  __bf16* WoTHi  = (__bf16*)(w + alloc((size_t)H_DIM * NOBS * 2));
  __bf16* WoTLo  = (__bf16*)(w + alloc((size_t)H_DIM * NOBS * 2));
  __bf16* pHi    = (__bf16*)(w + alloc((size_t)n * 2));
  __bf16* pLo    = (__bf16*)(w + alloc((size_t)n * 2));
  __bf16* cHi    = (__bf16*)(w + alloc((size_t)n * 2));
  __bf16* cLo    = (__bf16*)(w + alloc((size_t)n * 2));
  __bf16* jHi    = (__bf16*)(w + alloc((size_t)nBO * 2));
  __bf16* jLo    = (__bf16*)(w + alloc((size_t)nBO * 2));
  float*  part   = (float*)(w + alloc(256 * 4));
  int*    flags  = (int*)(w + alloc(8));

  auto blks = [](int m) { int b = (m + 255) / 256; return b > 4096 ? 4096 : b; };

  // ---- prep ----
  split_kernel <<<blks(H_DIM * H_DIM), 256, 0, stream>>>(Wr,   WrHi, WrLo, H_DIM * H_DIM);
  split_kernel <<<blks(NOBS * H_DIM),  256, 0, stream>>>(Wout, WoHi, WoLo, NOBS * H_DIM);
  splitT_kernel<<<blks(H_DIM * NOBS),  256, 0, stream>>>(Wout, WoTHi, WoTLo, H_DIM * NOBS);
  drive_kernel <<<blks(n), 256, 0, stream>>>(Win, dirs, drive, n);
  copy_kernel  <<<blks(n), 256, 0, stream>>>(ps, prevb, n);
  init_flags_kernel<<<1, 1, 0, stream>>>(flags);

  dim3 gWr(H_DIM / 64, B_DIM / 128);   // 16 x 32
  dim3 gLo(NOBS / 64,  B_DIM / 128);   //  2 x 32

  // cur = norm(relu(prev @ Wr^T + drive))
  split_kernel<<<blks(n), 256, 0, stream>>>(prevb, pHi, pLo, n);
  gemm_bf16x3<<<gWr, 256, 0, stream>>>(pHi, pLo, H_DIM, WrHi, WrLo, H_DIM, Rbuf, H_DIM, drive, 1, H_DIM);
  rownorm_kernel<<<B_DIM, 256, 0, stream>>>(Rbuf, curb);

  for (int it = 0; it < MAX_ITER; ++it) {
    split_kernel<<<blks(n), 256, 0, stream>>>(prevb, pHi, pLo, n);
    split_kernel<<<blks(n), 256, 0, stream>>>(curb,  cHi, cLo, n);
    reduce_diff_kernel<<<256, 256, 0, stream>>>(prevb, curb, part, n);
    flags_kernel<<<1, 256, 0, stream>>>(part, flags, 1.0f / (float)n);

    gemm_bf16x3<<<gLo, 256, 0, stream>>>(cHi, cLo, H_DIM, WoHi, WoLo, H_DIM, logits, NOBS, nullptr, 0, H_DIM);
    softmax_jv_kernel<<<B_DIM / 8, 256, 0, stream>>>(logits, obs, Jv, 1, nullptr);
    split_kernel<<<blks(nBO), 256, 0, stream>>>(Jv, jHi, jLo, nBO);
    gemm_bf16x3<<<gWr, 256, 0, stream>>>(jHi, jLo, NOBS, WoTHi, WoTLo, NOBS, Ubuf, H_DIM, nullptr, 0, NOBS);
    gemm_bf16x3<<<gWr, 256, 0, stream>>>(pHi, pLo, H_DIM, WrHi, WrLo, H_DIM, Rbuf, H_DIM, drive, 1, H_DIM);
    combine_norm_kernel<<<B_DIM, 256, 0, stream>>>(curb, Rbuf, Ubuf, newc);
    advance_kernel<<<4096, 256, 0, stream>>>(prevb, curb, newc, flags, n);
  }

  // outputs: preds = softmax(cur @ Wout^T), then cur
  split_kernel<<<blks(n), 256, 0, stream>>>(curb, cHi, cLo, n);
  gemm_bf16x3<<<gLo, 256, 0, stream>>>(cHi, cLo, H_DIM, WoHi, WoLo, H_DIM, logits, NOBS, nullptr, 0, H_DIM);
  softmax_jv_kernel<<<B_DIM / 8, 256, 0, stream>>>(logits, nullptr, nullptr, 0, out);
  copy_kernel<<<blks(n), 256, 0, stream>>>(curb, out + (size_t)B_DIM * NOBS, n);
}
